// LPN_62569083568221
// MI455X (gfx1250) — compile-verified, matrix operand order checked by reading, not run on hardware
//
#include <hip/hip_runtime.h>

// ---------------------------------------------------------------------------
// Types for CDNA5 WMMA (wave32): bf16 A/B fragments, f32 accumulator
// ---------------------------------------------------------------------------
typedef __attribute__((ext_vector_type(16))) __bf16 v16bf;
typedef __attribute__((ext_vector_type(8)))  float  v8f;

union FragBF { uint4 u[2]; v16bf v; };

__device__ __forceinline__ unsigned short f2bf(float f) {
    unsigned u = __float_as_uint(f);
    unsigned r = u + 0x7FFFu + ((u >> 16) & 1u);   // round-to-nearest-even
    return (unsigned short)(r >> 16);
}
__device__ __forceinline__ float bf2f(unsigned short h) {
    return __uint_as_float(((unsigned)h) << 16);
}

// Low 32 bits of a flat LDS pointer == LDS segment offset (ISA §10.2).
// The ptrtoint also captures the LDS object so the async-copy inline asm
// (with its "memory" clobber) is a legal writer of it.
__device__ __forceinline__ unsigned lds_off(const void* p) {
    return (unsigned)(unsigned long long)p;
}

// ---------------------------------------------------------------------------
// Pad + convert: feat f32 [8][H][W][C] -> bf16 [8][H+2][W+2][C], zero borders
// ---------------------------------------------------------------------------
__global__ __launch_bounds__(256)
void k_pad_f32_to_bf16(const float* __restrict__ x, unsigned short* __restrict__ y,
                       int H, int W, int C)
{
    const int n = 8 * (H + 2) * (W + 2) * C;
    int i = blockIdx.x * 256 + threadIdx.x;
    if (i >= n) return;
    int c  = i % C;
    int r  = i / C;
    int wp = r % (W + 2);
    int r2 = r / (W + 2);
    int hp = r2 % (H + 2);
    int d  = r2 / (H + 2);
    int h = hp - 1, w = wp - 1;
    unsigned short v = 0;
    if (h >= 0 && h < H && w >= 0 && w < W)
        v = f2bf(x[((size_t)(d * H + h) * W + w) * C + c]);
    y[i] = v;
}

// ---------------------------------------------------------------------------
// Weight convert + transpose: w f32 [3][3][Cin][192] -> bf16 [9][192][Cin]
// ---------------------------------------------------------------------------
__global__ __launch_bounds__(256)
void k_wgt_to_bf16_T(const float* __restrict__ wf, unsigned short* __restrict__ wt, int Cin)
{
    const int n = 9 * 192 * Cin;
    int i = blockIdx.x * 256 + threadIdx.x;
    if (i >= n) return;
    int ci = i % Cin;
    int r  = i / Cin;
    int nc = r % 192;
    int kp = r / 192;
    wt[i] = f2bf(wf[((size_t)kp * Cin + ci) * 192 + nc]);
}

// ---------------------------------------------------------------------------
// 3x3 SAME conv as implicit GEMM on WMMA with async double-buffered staging.
//   act : padded [8][H+2][W+2][Cin] bf16
//   wgt : [9][192][Cin] bf16 (B^T)
//   out : [8*H*W][192] f32
// Block = 256 threads (8 waves). Tile: 128 (M) x 64 (N). Wave: 32x32 (4 WMMA).
// ---------------------------------------------------------------------------
#define LDAB 40   // padded LDS row stride (bf16 units); 80B, 16B-aligned

__global__ __launch_bounds__(256)
void k_conv3x3_wmma(const unsigned short* __restrict__ act,
                    const unsigned short* __restrict__ wgt,
                    float* __restrict__ out,
                    int H, int W, int Cin)
{
    const int Cout   = 192;
    const int blockM = blockIdx.x * 128;
    const int nBase  = blockIdx.y * 64;
    const int Wp     = W + 2;

    // double-buffered tiles: [set][A 128 rows | B 64 rows]
    __shared__ __align__(16) unsigned short smem[2][(128 + 64) * LDAB];

    const int t    = threadIdx.x;
    const int lane = t & 31;
    const int wave = t >> 5;
    const int wm   = wave & 3;     // 4 M quarters of 32 rows
    const int wn   = wave >> 2;    // 2 N halves of 32 cols
    const int lrow = lane & 15;
    const int hi   = lane >> 4;    // K sub-range select per ISA 16-bit striping

    v8f acc00 = {}, acc01 = {}, acc10 = {}, acc11 = {};

    // A staging: thread stages rows (t>>2) and (t>>2)+64, channels akk..akk+7
    const int arow = t >> 2;
    const int akk  = (t & 3) * 8;
    int pbase0, pbase1;
    {
        int p  = blockM + arow;
        int d  = p / (H * W);
        int hw = p - d * H * W;
        int h  = hw / W;
        int w  = hw - h * W;
        pbase0 = ((d * (H + 2) + h) * Wp + w) * Cin;
        p  = blockM + arow + 64;
        d  = p / (H * W);
        hw = p - d * H * W;
        h  = hw / W;
        w  = hw - h * W;
        pbase1 = ((d * (H + 2) + h) * Wp + w) * Cin;
    }
    // B staging: thread stages n = t>>2, channels akk..akk+7
    const int bn = t >> 2;

    // LDS byte offsets of this thread's staging slots (captures smem!)
    unsigned ao0[2], ao1[2], bo[2];
    #pragma unroll
    for (int s = 0; s < 2; ++s) {
        ao0[s] = lds_off(&smem[s][(arow)      * LDAB + akk]);
        ao1[s] = lds_off(&smem[s][(arow + 64) * LDAB + akk]);
        bo[s]  = lds_off(&smem[s][(128 + bn)  * LDAB + akk]);
    }

    const int nc    = Cin >> 5;     // 32-channel chunks per tap
    const int KITER = 9 * nc;

    auto issue = [&](int j, int s) {
        int kp = j / nc;
        int cb = (j - kp * nc) << 5;
        int ky = kp / 3, kx = kp - ky * 3;
        int tap = (ky * Wp + kx) * Cin + cb + akk;
        const unsigned short* a0 = act + (size_t)(pbase0 + tap);
        const unsigned short* a1 = act + (size_t)(pbase1 + tap);
        const unsigned short* b0 = wgt + ((size_t)(kp * 192 + nBase + bn)) * Cin
                                       + (size_t)(cb + akk);
        asm volatile("global_load_async_to_lds_b128 %0, %1, off"
                     :: "v"(ao0[s]), "v"((unsigned long long)a0) : "memory");
        asm volatile("global_load_async_to_lds_b128 %0, %1, off"
                     :: "v"(ao1[s]), "v"((unsigned long long)a1) : "memory");
        asm volatile("global_load_async_to_lds_b128 %0, %1, off"
                     :: "v"(bo[s]),  "v"((unsigned long long)b0) : "memory");
    };

    issue(0, 0);                    // prologue: fill set 0
    int sel = 0;

    for (int kc = 0; kc < KITER; ++kc) {
        int nj = (kc + 1 < KITER) ? kc + 1 : KITER - 1;   // tail: harmless re-issue
        issue(nj, sel ^ 1);
        // wait until only the 3 just-issued copies are outstanding -> set `sel` done
        asm volatile("s_wait_asynccnt 0x3" ::: "memory");
        __syncthreads();

        const unsigned short* lA = &smem[sel][0];
        const unsigned short* lB = &smem[sel][128 * LDAB];

        FragBF fa0, fa1, fb0, fb1;
        const unsigned short* pa0 = &lA[(wm * 32 + lrow) * LDAB + hi * 8];
        fa0.u[0] = *(const uint4*)(pa0);
        fa0.u[1] = *(const uint4*)(pa0 + 16);
        const unsigned short* pa1 = &lA[(wm * 32 + 16 + lrow) * LDAB + hi * 8];
        fa1.u[0] = *(const uint4*)(pa1);
        fa1.u[1] = *(const uint4*)(pa1 + 16);
        const unsigned short* pb0 = &lB[(wn * 32 + lrow) * LDAB + hi * 8];
        fb0.u[0] = *(const uint4*)(pb0);
        fb0.u[1] = *(const uint4*)(pb0 + 16);
        const unsigned short* pb1 = &lB[(wn * 32 + 16 + lrow) * LDAB + hi * 8];
        fb1.u[0] = *(const uint4*)(pb1);
        fb1.u[1] = *(const uint4*)(pb1 + 16);

        acc00 = __builtin_amdgcn_wmma_f32_16x16x32_bf16(
                    false, fa0.v, false, fb0.v, (short)0, acc00, false, false);
        acc01 = __builtin_amdgcn_wmma_f32_16x16x32_bf16(
                    false, fa0.v, false, fb1.v, (short)0, acc01, false, false);
        acc10 = __builtin_amdgcn_wmma_f32_16x16x32_bf16(
                    false, fa1.v, false, fb0.v, (short)0, acc10, false, false);
        acc11 = __builtin_amdgcn_wmma_f32_16x16x32_bf16(
                    false, fa1.v, false, fb1.v, (short)0, acc11, false, false);

        __syncthreads();
        sel ^= 1;
    }

    // epilogue: C/D layout (VGPR r: lanes 0-15 -> M=r, lanes 16-31 -> M=r+8)
    const int m0 = blockM + wm * 32 + hi * 8;
    const int n0 = nBase + wn * 32 + lrow;
    #pragma unroll
    for (int r = 0; r < 8; ++r) {
        out[(size_t)(m0 + r)      * Cout + n0]      = acc00[r];
        out[(size_t)(m0 + r)      * Cout + n0 + 16] = acc01[r];
        out[(size_t)(m0 + 16 + r) * Cout + n0]      = acc10[r];
        out[(size_t)(m0 + 16 + r) * Cout + n0 + 16] = acc11[r];
    }
}

// ---------------------------------------------------------------------------
// GroupNorm (num_groups == channels): per-channel sum / sumsq over all rows
// ---------------------------------------------------------------------------
__global__ __launch_bounds__(256)
void k_gn_stats(const float* __restrict__ x, float* __restrict__ sums,
                float* __restrict__ sumsq, int M)
{
    const int c = blockIdx.x;  // 0..191
    float s = 0.f, s2 = 0.f;
    for (int i = threadIdx.x; i < M; i += 256) {
        float v = x[(size_t)i * 192 + c];
        s += v; s2 += v * v;
    }
    __shared__ float rs[256], r2[256];
    rs[threadIdx.x] = s; r2[threadIdx.x] = s2;
    __syncthreads();
    for (int st = 128; st > 0; st >>= 1) {
        if (threadIdx.x < st) {
            rs[threadIdx.x] += rs[threadIdx.x + st];
            r2[threadIdx.x] += r2[threadIdx.x + st];
        }
        __syncthreads();
    }
    if (threadIdx.x == 0) { sums[c] = rs[0]; sumsq[c] = r2[0]; }
}

__global__ void k_gn_final(const float* __restrict__ sums, const float* __restrict__ sumsq,
                           const float* __restrict__ scale, const float* __restrict__ bias,
                           float* __restrict__ ga, float* __restrict__ gb, int M)
{
    int c = threadIdx.x;
    if (c >= 192) return;
    float inv  = 1.0f / (float)M;
    float mean = sums[c] * inv;
    float var  = sumsq[c] * inv - mean * mean;
    float a    = rsqrtf(var + 1e-6f) * scale[c];
    ga[c] = a;
    gb[c] = bias[c] - mean * a;
}

// normalize + ReLU + bf16 convert, written into PADDED layout (zero borders)
__global__ __launch_bounds__(256)
void k_gn_relu_pad(const float* __restrict__ x, const float* __restrict__ ga,
                   const float* __restrict__ gb, unsigned short* __restrict__ y,
                   int H, int W)
{
    const int C = 192;
    const int n = 8 * (H + 2) * (W + 2) * C;
    int i = blockIdx.x * 256 + threadIdx.x;
    if (i >= n) return;
    int c  = i % C;
    int r  = i / C;
    int wp = r % (W + 2);
    int r2 = r / (W + 2);
    int hp = r2 % (H + 2);
    int d  = r2 / (H + 2);
    int h = hp - 1, w = wp - 1;
    unsigned short v = 0;
    if (h >= 0 && h < H && w >= 0 && w < W) {
        float t = x[((size_t)(d * H + h) * W + w) * C + c];
        v = f2bf(fmaxf(0.f, fmaf(t, ga[c], gb[c])));
    }
    y[i] = v;
}

// ---------------------------------------------------------------------------
// Heads: 1x1 cls(2)/regr(3) + softmax score + grid locations + validity
// ---------------------------------------------------------------------------
__global__ __launch_bounds__(256)
void k_head(const unsigned short* __restrict__ act,
            const float* __restrict__ clsw, const float* __restrict__ clsb,
            const float* __restrict__ regw, const float* __restrict__ regb,
            float* __restrict__ scores, float* __restrict__ locs,
            float* __restrict__ clsbuf,
            int H, int W, float scale, int off)
{
    const int M = 8 * H * W;
    int p = blockIdx.x * 256 + threadIdx.x;
    if (p >= M) return;
    int d  = p / (H * W);
    int hw = p - d * H * W;
    int h  = hw / W;
    int w  = hw - h * W;
    const unsigned short* a =
        act + ((size_t)(d * (H + 2) + h + 1) * (W + 2) + w + 1) * 192;

    float c0 = clsb[0], c1 = clsb[1];
    float r0 = regb[0], r1 = regb[1], r2 = regb[2];
    for (int c = 0; c < 192; ++c) {
        float x = bf2f(a[c]);
        c0 = fmaf(x, clsw[c * 2 + 0], c0);
        c1 = fmaf(x, clsw[c * 2 + 1], c1);
        r0 = fmaf(x, regw[c * 3 + 0], r0);
        r1 = fmaf(x, regw[c * 3 + 1], r1);
        r2 = fmaf(x, regw[c * 3 + 2], r2);
    }
    float mx = fmaxf(c0, c1);
    float e0 = __expf(c0 - mx), e1 = __expf(c1 - mx);
    float sc = e0 / (e0 + e1);

    float lz = (d + 0.5f) * 5.0f  + r0;
    float ly = (h + 0.5f) * scale + r1;
    float lx = (w + 0.5f) * scale + r2;
    bool valid = (lz > 0.f) && (ly > 0.f) && (lx > 0.f) &&
                 (lz < 40.0f) && (ly < H * scale) && (lx < W * scale);

    scores[off + p] = valid ? sc : -1.0f;
    locs[(size_t)(off + p) * 3 + 0] = lz;
    locs[(size_t)(off + p) * 3 + 1] = ly;
    locs[(size_t)(off + p) * 3 + 2] = lx;
    clsbuf[off + p] = 0.0f;   // N_CLS == 1 -> argmax over one class is 0
}

// ---------------------------------------------------------------------------
// Sequential distance-NMS: 1280 global-argmax rounds (first-index tie-break).
// Writes d_out = [scores 1280][locations 1280x3 (z already /5)][classes 1280]
// ---------------------------------------------------------------------------
__global__ __launch_bounds__(1024)
void k_nms(float* __restrict__ scores, const float* __restrict__ locs,
           const float* __restrict__ clsbuf, float* __restrict__ d_out)
{
    const int   N    = 43008;
    const int   ITER = 1280;
    const float THR  = 1.0f / 64.0f;   // 1 / NMS_THRESHOLD^2
    const int   tid  = threadIdx.x;

    __shared__ float ss[1024];
    __shared__ int   si[1024];
    __shared__ float kz, ky, kx;
    __shared__ int   mcnt;
    if (tid == 0) mcnt = 0;
    __syncthreads();

    for (int it = 0; it < ITER; ++it) {
        float best = -INFINITY;
        int   bi   = N;
        for (int i = tid; i < N; i += 1024) {
            float s = scores[i];
            if (s > best || (s == best && i < bi)) { best = s; bi = i; }
        }
        ss[tid] = best; si[tid] = bi;
        __syncthreads();
        for (int st = 512; st > 0; st >>= 1) {
            if (tid < st) {
                float so = ss[tid + st]; int io = si[tid + st];
                if (so > ss[tid] || (so == ss[tid] && io < si[tid])) {
                    ss[tid] = so; si[tid] = io;
                }
            }
            __syncthreads();
        }
        if (tid == 0) {
            int   k = si[0];
            float s = ss[0];
            float lz = locs[(size_t)k * 3 + 0];
            float ly = locs[(size_t)k * 3 + 1];
            float lx = locs[(size_t)k * 3 + 2];
            kz = lz; ky = ly; kx = lx;
            bool valid = (s >= 0.2f);
            d_out[it] = valid ? s : -1.0f;
            float oz = valid ? lz : -1.0f;
            float oy = valid ? ly : -1.0f;
            float ox = valid ? lx : -1.0f;
            d_out[1280 + (size_t)it * 3 + 0] = oz * 0.2f;  // / Z_SCALE (also -1 -> -0.2)
            d_out[1280 + (size_t)it * 3 + 1] = oy;
            d_out[1280 + (size_t)it * 3 + 2] = ox;
            if (valid) { d_out[5120 + mcnt] = clsbuf[k]; mcnt++; }
        }
        __syncthreads();
        float pz = kz, py = ky, px = kx;
        for (int i = tid; i < N; i += 1024) {
            float dz = locs[(size_t)i * 3 + 0] - pz;
            float dy = locs[(size_t)i * 3 + 1] - py;
            float dx = locs[(size_t)i * 3 + 2] - px;
            float d2 = dz * dz + dy * dy + dx * dx;
            if (d2 * THR < 1.0f) scores[i] = -INFINITY;
        }
        __syncthreads();
    }
    for (int i = mcnt + tid; i < ITER; i += 1024)
        d_out[5120 + i] = -1.0f;
}

// ---------------------------------------------------------------------------
// Orchestration
// ---------------------------------------------------------------------------
extern "C" void kernel_launch(void* const* d_in, const int* in_sizes, int n_in,
                              void* d_out, int out_size, void* d_ws, size_t ws_size,
                              hipStream_t stream)
{
    (void)in_sizes; (void)n_in; (void)out_size; (void)ws_size;

    const float* feats[3] = {
        (const float*)d_in[0], (const float*)d_in[1], (const float*)d_in[2]
    };

    // workspace layout (all offsets 16B-aligned); sized for level 0 (H=W=64)
    char* ws = (char*)d_ws;
    size_t o = 0;
    unsigned short* actA  = (unsigned short*)(ws + o); o += 8ull * 66 * 66 * 256 * 2; // padded in
    unsigned short* actP0 = (unsigned short*)(ws + o); o += 8ull * 66 * 66 * 192 * 2; // padded
    unsigned short* actP1 = (unsigned short*)(ws + o); o += 8ull * 66 * 66 * 192 * 2; // padded
    float*          convo = (float*)(ws + o);          o += 32768ull * 192 * 4;       // unpadded
    unsigned short* wbf   = (unsigned short*)(ws + o); o += 9ull * 192 * 256 * 2;     // B^T
    float*          gsum  = (float*)(ws + o);          o += 192 * 4;
    float*          gsq   = (float*)(ws + o);          o += 192 * 4;
    float*          ga    = (float*)(ws + o);          o += 192 * 4;
    float*          gb    = (float*)(ws + o);          o += 192 * 4;
    float*          scr   = (float*)(ws + o);          o += 43008ull * 4;
    float*          lcs   = (float*)(ws + o);          o += 43008ull * 12;
    float*          clb   = (float*)(ws + o);          o += 43008ull * 4;

    const int   Hs[3]   = {64, 32, 16};
    const float scl[3]  = {4.f, 8.f, 16.f};
    const int   offs[3] = {0, 32768, 32768 + 8192};

    for (int l = 0; l < 3; ++l) {
        const int H = Hs[l], W = Hs[l];
        const int M = 8 * H * W;
        const int npad = 8 * (H + 2) * (W + 2);
        const int pbase = 3 + l * 16;

        k_pad_f32_to_bf16<<<(npad * 256 + 255) / 256, 256, 0, stream>>>(
            feats[l], actA, H, W, 256);

        const unsigned short* in = actA;
        int Cin = 256;
        unsigned short* outs[4] = {actP0, actP1, actP0, actP1};

        for (int i = 0; i < 4; ++i) {
            const float* wf  = (const float*)d_in[pbase + i * 3 + 0];
            const float* gns = (const float*)d_in[pbase + i * 3 + 1];
            const float* gnb = (const float*)d_in[pbase + i * 3 + 2];

            int wcount = 9 * 192 * Cin;
            k_wgt_to_bf16_T<<<(wcount + 255) / 256, 256, 0, stream>>>(wf, wbf, Cin);

            dim3 grid(M / 128, 3);
            k_conv3x3_wmma<<<grid, 256, 0, stream>>>(in, wbf, convo, H, W, Cin);

            k_gn_stats<<<192, 256, 0, stream>>>(convo, gsum, gsq, M);
            k_gn_final<<<1, 192, 0, stream>>>(gsum, gsq, gns, gnb, ga, gb, M);
            k_gn_relu_pad<<<(npad * 192 + 255) / 256, 256, 0, stream>>>(
                convo, ga, gb, outs[i], H, W);

            in = outs[i];
            Cin = 192;
        }

        const float* clsw = (const float*)d_in[pbase + 12];
        const float* clsb = (const float*)d_in[pbase + 13];
        const float* regw = (const float*)d_in[pbase + 14];
        const float* regb = (const float*)d_in[pbase + 15];
        k_head<<<(M + 255) / 256, 256, 0, stream>>>(in, clsw, clsb, regw, regb,
                                                    scr, lcs, clb, H, W, scl[l], offs[l]);
    }

    k_nms<<<1, 1024, 0, stream>>>(scr, lcs, clb, (float*)d_out);
}
